// CombinedLoss_5394478924576
// MI455X (gfx1250) — matrix operand order repaired and checked by reading, not compiled, not run on hardware
//
#include <hip/hip_runtime.h>
#include <hip/hip_bf16.h>
#include <stdint.h>

// ---------------------------------------------------------------------------
// CombinedLoss for MI455X (gfx1250): MSE + 0.5*SSIM + 0.1*perceptual
// Conv stack (~620 GFLOP) as implicit-GEMM with v_wmma_f32_16x16x32_f16,
// tiles staged via CDNA5 async global->LDS (ASYNCcnt) instructions.
// ---------------------------------------------------------------------------

typedef __attribute__((ext_vector_type(16))) _Float16 v16h;
typedef __attribute__((ext_vector_type(8)))  float    v8f;

#define BATCH 8
#define HH1 256
#define WW1 256
#define CC1 64
#define CC2 128
#define HH2 128
#define WW2 128
#define CC3 256
#define CC4 256
#define SSIM_OW 250   // 256 - 7 + 1

union FragH { uint4 q[2]; v16h v; };

// Async copy 16B global -> LDS (gfx1250 GLOBAL_LOAD_ASYNC_TO_LDS_B128).
// ldsoff = LDS byte address (low 32 bits of generic shared pointer: the LDS
// aperture keeps the offset in addr[31:0], ISA 10.2). Tracked by ASYNCcnt.
__device__ __forceinline__ void async_ld_b128(uint32_t ldsoff, const void* gaddr) {
    asm volatile("global_load_async_to_lds_b128 %0, %1, off"
                 :: "v"(ldsoff), "v"(gaddr) : "memory");
}
__device__ __forceinline__ void wait_async0() {
    asm volatile("s_wait_asynccnt 0" ::: "memory");
}

// ---------------------------------------------------------------------------
// Zero the accumulator slots (mse, ssim, perceptual) each launch.
// ---------------------------------------------------------------------------
__global__ void zero_accum(float* a) {
    if (threadIdx.x < 8) a[threadIdx.x] = 0.0f;
}

// ---------------------------------------------------------------------------
// Weight transform: w[Cout][Cin][3][3] f32 -> tiled f16 layout matching the
// conv kernel's LDS B-tile exactly:
//   wt[ (kblk*(Cout/64) + cblk)*2048 + n*32 + kk ]
// where k = (3*ky+kx)*Cin + ci = kblk*32 + kk, co = cblk*64 + n.
// B staging then becomes one contiguous b128 per thread.
// ---------------------------------------------------------------------------
__global__ void wt_transform(const float* __restrict__ w, _Float16* __restrict__ wt,
                             int Cin, int Cout) {
    int i = blockIdx.x * blockDim.x + threadIdx.x;
    int total = Cin * 9 * Cout;
    if (i >= total) return;
    int co  = i % Cout;
    int k   = i / Cout;
    int tap = k / Cin;
    int ci  = k - tap * Cin;
    float v = w[(size_t)(co * Cin + ci) * 9 + tap];
    int kblk = k >> 5, kk = k & 31;
    int cblk = co >> 6, n  = co & 63;
    int Cout64 = Cout >> 6;
    wt[(((size_t)kblk * Cout64 + cblk) << 11) + (n << 5) + kk] = (_Float16)v;
}

// ---------------------------------------------------------------------------
// conv1: 1 -> 64 channels, 3x3 SAME, direct (K=9, too small for WMMA).
// Input (B,H,W) f32, output NHWC f16. One thread per pixel.
// ---------------------------------------------------------------------------
__global__ __launch_bounds__(256)
void conv1_direct(const float* __restrict__ in, const float* __restrict__ w1,
                  const float* __restrict__ b1, _Float16* __restrict__ out) {
    __shared__ float sw[64 * 9];
    __shared__ float sb[64];
    for (int i = threadIdx.x; i < 64 * 9; i += 256) sw[i] = w1[i];
    if (threadIdx.x < 64) sb[threadIdx.x] = b1[threadIdx.x];
    __syncthreads();

    int p = blockIdx.x * 256 + threadIdx.x;           // exact grid, no tail
    int x = p % WW1;
    int y = (p / WW1) % HH1;
    int b = p / (WW1 * HH1);

    float patch[9];
#pragma unroll
    for (int t = 0; t < 9; ++t) {
        int yy = y + t / 3 - 1;
        int xx = x + t % 3 - 1;
        bool ok = (yy >= 0) && (yy < HH1) && (xx >= 0) && (xx < WW1);
        patch[t] = ok ? in[((size_t)b * HH1 + yy) * WW1 + xx] : 0.0f;
    }
    _Float16* o = out + (size_t)p * CC1;
#pragma unroll 4
    for (int co = 0; co < 64; ++co) {
        float acc = sb[co];
#pragma unroll
        for (int t = 0; t < 9; ++t) acc += patch[t] * sw[co * 9 + t];
        o[co] = (_Float16)(acc > 0.0f ? acc : 0.0f);
    }
}

// ---------------------------------------------------------------------------
// WMMA implicit-GEMM 3x3 SAME conv, NHWC f16 in, K-order = (ky,kx,ci).
// Block: 256 threads (8 wave32), tile 64 pixels x 64 cout.
// Wave w: M-sub = (w&3)*16 pixels, N-sub = (w>>2)*32 channels (2 accums).
// Tiles staged with global_load_async_to_lds_b128 (ASYNCcnt).
// MODE 0: relu -> f16 NHWC.  MODE 1: relu -> f32 (sr features).
// MODE 2: relu, diff vs stored f32 sr features, sum of squares -> accum.
// ---------------------------------------------------------------------------
template <int MODE>
__global__ __launch_bounds__(256)
void conv3x3_wmma(const _Float16* __restrict__ act, const _Float16* __restrict__ wt,
                  const float* __restrict__ bias,
                  _Float16* __restrict__ outh, float* __restrict__ outf,
                  float* __restrict__ accum,
                  int H, int W, int Cin, int Cout) {
    __shared__ __align__(16) _Float16 sA[64 * 32];   // [pixel][k]  4 KB
    __shared__ __align__(16) _Float16 sB[64 * 32];   // [n][k]      4 KB
    __shared__ float sred[256];

    const int p0  = blockIdx.x * 64;                 // strip start (one row: 64|W)
    const int hw  = H * W;
    const int b   = p0 / hw;
    const int rem = p0 - b * hw;
    const int y   = rem / W;
    const int x0  = rem - y * W;
    const int cb  = blockIdx.y;                      // cout block (64 wide)
    const int co0 = cb << 6;
    const int Cout64 = Cout >> 6;

    const int tid  = threadIdx.x;
    const int lane = tid & 31;
    const int wv   = tid >> 5;
    const int m0   = (wv & 3) * 16;
    const int n0   = (wv >> 2) * 32;

    // ---- staging addresses (constant across K-steps) ----
    const int pixS  = tid >> 2;                      // pixel this thread stages
    const int quadS = (tid & 3) * 8;                 // channel sub-offset (f16)
    const uint32_t ldsA = (uint32_t)(size_t)&sA[pixS * 32 + quadS];
    const uint32_t ldsB = (uint32_t)(size_t)&sB[tid * 8];
    const int xxS = x0 + pixS;                       // before +kx

    // ---- fragment LDS pointers (lane-constant across K-steps) ----
    const int klo  = (lane >> 4) * 16;
    const int pixF = m0 + (lane & 15);
    const uint4* fa0p = (const uint4*)&sA[pixF * 32 + klo];
    const uint4* fa1p = (const uint4*)&sA[pixF * 32 + klo + 8];
    const int koff = (lane >> 4) * 16;
    const int rB0  = n0 + (lane & 15);
    const uint4* fb00p = (const uint4*)&sB[rB0 * 32 + koff];
    const uint4* fb01p = (const uint4*)&sB[rB0 * 32 + koff + 8];
    const uint4* fb10p = (const uint4*)&sB[(rB0 + 16) * 32 + koff];
    const uint4* fb11p = (const uint4*)&sB[(rB0 + 16) * 32 + koff + 8];

    v8f acc0 = {};
    v8f acc1 = {};

    for (int tap = 0; tap < 9; ++tap) {
        const int ky = tap / 3 - 1;
        const int kx = tap % 3 - 1;
        const int yy = y + ky;
        const int xx = xxS + kx;
        const bool okA = (yy >= 0) && (yy < H) && (xx >= 0) && (xx < W);
        const _Float16* rowA =
            act + (((size_t)(b * H + yy) * W + xx) * Cin + quadS);
        const int kblkBase = (tap * Cin) >> 5;

        for (int ci0 = 0; ci0 < Cin; ci0 += 32) {
            // ---- stage A: 64 pixels x 32 ch; async b128, zeros for padding ----
            if (okA) {
                async_ld_b128(ldsA, rowA + ci0);
            } else {
                uint4 z = {0u, 0u, 0u, 0u};
                *(uint4*)&sA[pixS * 32 + quadS] = z;
            }
            // ---- stage B: contiguous pre-tiled weights, one async b128 ----
            {
                const int kblk = kblkBase + (ci0 >> 5);
                const _Float16* srcB =
                    wt + (((size_t)(kblk * Cout64 + cb)) << 11) + tid * 8;
                async_ld_b128(ldsB, srcB);
            }
            wait_async0();
            __syncthreads();

            // ---- fragments (k = 16*(lane>>4) + s, same order in A and B) ----
            FragH fa, fb0, fb1;
            fa.q[0]  = *fa0p;  fa.q[1]  = *fa1p;
            fb0.q[0] = *fb00p; fb0.q[1] = *fb01p;
            fb1.q[0] = *fb10p; fb1.q[1] = *fb11p;

            acc0 = __builtin_amdgcn_wmma_f32_16x16x32_f16(
                false, fa.v, false, fb0.v, (short)0, acc0, false, false);
            acc1 = __builtin_amdgcn_wmma_f32_16x16x32_f16(
                false, fa.v, false, fb1.v, (short)0, acc1, false, false);
            __syncthreads();
        }
    }

    // ---- epilogue: C/D layout M = r + 8*(lane>=16), N = lane&15 ----
    float psum = 0.0f;
#pragma unroll
    for (int r = 0; r < 8; ++r) {
        const int pix = m0 + (lane >> 4) * 8 + r;
        const int x   = x0 + pix;
        const size_t base = ((size_t)(b * H + y) * W + x) * Cout;
#pragma unroll
        for (int j = 0; j < 2; ++j) {
            const int co = co0 + n0 + j * 16 + (lane & 15);
            float v = (j ? acc1[r] : acc0[r]) + bias[co];
            v = v > 0.0f ? v : 0.0f;
            if (MODE == 0) {
                outh[base + co] = (_Float16)v;
            } else if (MODE == 1) {
                outf[base + co] = v;
            } else {
                float d = v - outf[base + co];
                psum += d * d;
            }
        }
    }
    if (MODE == 2) {
        sred[tid] = psum;
        __syncthreads();
        for (int s = 128; s > 0; s >>= 1) {
            if (tid < s) sred[tid] += sred[tid + s];
            __syncthreads();
        }
        if (tid == 0) atomicAdd(accum, sred[0]);
    }
}

// ---------------------------------------------------------------------------
// 2x2 maxpool, NHWC f16: (B,256,256,C) -> (B,128,128,C)
// ---------------------------------------------------------------------------
__global__ __launch_bounds__(256)
void maxpool2x2(const _Float16* __restrict__ in, _Float16* __restrict__ out, int C) {
    size_t i = (size_t)blockIdx.x * 256 + threadIdx.x;
    size_t total = (size_t)BATCH * HH2 * WW2 * C;
    if (i >= total) return;
    int c = (int)(i % C);
    size_t t = i / C;
    int x = (int)(t % WW2); t /= WW2;
    int y = (int)(t % HH2);
    int b = (int)(t / HH2);
    const _Float16* p = in + (((size_t)(b * HH1 + 2 * y) * WW1 + 2 * x) * C + c);
    const size_t rs = (size_t)WW1 * C;
    float m = (float)p[0];
    m = fmaxf(m, (float)p[C]);
    m = fmaxf(m, (float)p[rs]);
    m = fmaxf(m, (float)p[rs + C]);
    out[i] = (_Float16)m;
}

// ---------------------------------------------------------------------------
// MSE partial sum -> accum[0]
// ---------------------------------------------------------------------------
__global__ __launch_bounds__(256)
void mse_kernel(const float* __restrict__ a, const float* __restrict__ b,
                float* __restrict__ accum, int N) {
    __shared__ float sred[256];
    float s = 0.0f;
    for (int i = blockIdx.x * 256 + threadIdx.x; i < N; i += gridDim.x * 256) {
        float d = a[i] - b[i];
        s += d * d;
    }
    sred[threadIdx.x] = s;
    __syncthreads();
    for (int sft = 128; sft > 0; sft >>= 1) {
        if (threadIdx.x < sft) sred[threadIdx.x] += sred[threadIdx.x + sft];
        __syncthreads();
    }
    if (threadIdx.x == 0) atomicAdd(accum + 0, sred[0]);
}

// ---------------------------------------------------------------------------
// SSIM (skimage-style, 7x7 VALID uniform window). One thread per window;
// accumulates sum(S) into accum[1].
// ---------------------------------------------------------------------------
__global__ __launch_bounds__(256)
void ssim_kernel(const float* __restrict__ X, const float* __restrict__ Y,
                 float* __restrict__ accum) {
    __shared__ float sred[256];
    const int total = BATCH * SSIM_OW * SSIM_OW;
    int idx = blockIdx.x * 256 + threadIdx.x;
    float S = 0.0f;
    if (idx < total) {
        int xo = idx % SSIM_OW;
        int t  = idx / SSIM_OW;
        int yo = t % SSIM_OW;
        int b  = t / SSIM_OW;
        const float* px = X + (size_t)b * HH1 * WW1 + (size_t)yo * WW1 + xo;
        const float* py = Y + (size_t)b * HH1 * WW1 + (size_t)yo * WW1 + xo;
        float sx = 0.f, sy = 0.f, sxx = 0.f, syy = 0.f, sxy = 0.f;
#pragma unroll
        for (int i = 0; i < 7; ++i) {
#pragma unroll
            for (int j = 0; j < 7; ++j) {
                float a = px[i * WW1 + j];
                float c = py[i * WW1 + j];
                sx += a; sy += c;
                sxx += a * a; syy += c * c; sxy += a * c;
            }
        }
        const float inv = 1.0f / 49.0f;
        const float cov_norm = 49.0f / 48.0f;
        float ux = sx * inv, uy = sy * inv;
        float vx  = cov_norm * (sxx * inv - ux * ux);
        float vy  = cov_norm * (syy * inv - uy * uy);
        float vxy = cov_norm * (sxy * inv - ux * uy);
        const float C1v = 0.01f * 0.01f;
        const float C2v = 0.03f * 0.03f;
        float A1 = 2.0f * ux * uy + C1v;
        float A2 = 2.0f * vxy + C2v;
        float B1 = ux * ux + uy * uy + C1v;
        float B2 = vx + vy + C2v;
        S = (A1 * A2) / (B1 * B2);
    }
    sred[threadIdx.x] = S;
    __syncthreads();
    for (int sft = 128; sft > 0; sft >>= 1) {
        if (threadIdx.x < sft) sred[threadIdx.x] += sred[threadIdx.x + sft];
        __syncthreads();
    }
    if (threadIdx.x == 0) atomicAdd(accum + 1, sred[0]);
}

// ---------------------------------------------------------------------------
// Final combine: out = {total, mse, ssim_l, perceptual}
// ---------------------------------------------------------------------------
__global__ void finalize(const float* __restrict__ accum, float* __restrict__ out) {
    if (threadIdx.x == 0 && blockIdx.x == 0) {
        float mse   = accum[0] / (float)(BATCH * HH1 * WW1);
        float ssimm = accum[1] / (float)(BATCH * SSIM_OW * SSIM_OW);
        float ssim_l = 1.0f - ssimm;
        float perc  = accum[2] / (float)((size_t)BATCH * CC4 * HH2 * WW2);
        out[0] = mse + 0.5f * ssim_l + 0.1f * perc;
        out[1] = mse;
        out[2] = ssim_l;
        out[3] = perc;
    }
}

// ---------------------------------------------------------------------------
// Host launcher
// ---------------------------------------------------------------------------
extern "C" void kernel_launch(void* const* d_in, const int* in_sizes, int n_in,
                              void* d_out, int out_size, void* d_ws, size_t ws_size,
                              hipStream_t stream) {
    (void)in_sizes; (void)n_in; (void)out_size; (void)ws_size;
    const float* sr = (const float*)d_in[0];
    const float* hr = (const float*)d_in[1];
    const float* w1 = (const float*)d_in[2];
    const float* b1 = (const float*)d_in[3];
    const float* w2 = (const float*)d_in[4];
    const float* b2 = (const float*)d_in[5];
    const float* w3 = (const float*)d_in[6];
    const float* b3 = (const float*)d_in[7];
    const float* w4 = (const float*)d_in[8];
    const float* b4 = (const float*)d_in[9];
    float* out = (float*)d_out;

    char* ws = (char*)d_ws;
    size_t off = 0;
    auto take = [&](size_t bytes) {
        void* p = ws + off;
        off += (bytes + 255) & ~(size_t)255;
        return p;
    };
    float*     accum = (float*)take(256);
    _Float16*  wt2   = (_Float16*)take((size_t)576 * 128 * 2);
    _Float16*  wt3   = (_Float16*)take((size_t)1152 * 256 * 2);
    _Float16*  wt4   = (_Float16*)take((size_t)2304 * 256 * 2);
    _Float16*  a1    = (_Float16*)take((size_t)BATCH * HH1 * WW1 * CC1 * 2);
    _Float16*  a2    = (_Float16*)take((size_t)BATCH * HH1 * WW1 * CC2 * 2);
    _Float16*  a2p   = (_Float16*)take((size_t)BATCH * HH2 * WW2 * CC2 * 2);
    _Float16*  a3    = (_Float16*)take((size_t)BATCH * HH2 * WW2 * CC3 * 2);
    float*     srf   = (float*)take((size_t)BATCH * HH2 * WW2 * CC4 * 4);

    zero_accum<<<1, 32, 0, stream>>>(accum);

    // Pre-tile conv weights into the conv kernel's LDS B layout (f16)
    wt_transform<<<(576 * 128 + 255) / 256, 256, 0, stream>>>(w2, wt2, 64, 128);
    wt_transform<<<(1152 * 256 + 255) / 256, 256, 0, stream>>>(w3, wt3, 128, 256);
    wt_transform<<<(2304 * 256 + 255) / 256, 256, 0, stream>>>(w4, wt4, 256, 256);

    // MSE + SSIM (independent of conv pipeline)
    const int Nmse = BATCH * HH1 * WW1;
    mse_kernel<<<1024, 256, 0, stream>>>(sr, hr, accum, Nmse);
    const int Nssim = BATCH * SSIM_OW * SSIM_OW;
    ssim_kernel<<<(Nssim + 255) / 256, 256, 0, stream>>>(sr, hr, accum);

    const dim3 gC2((BATCH * HH1 * WW1) / 64, CC2 / 64);   // 8192 x 2
    const dim3 gC3((BATCH * HH2 * WW2) / 64, CC3 / 64);   //  2048 x 4
    const dim3 gC4((BATCH * HH2 * WW2) / 64, CC4 / 64);   //  2048 x 4
    const int  gP  = (int)(((size_t)BATCH * HH2 * WW2 * CC2 + 255) / 256);
    const int  gC1 = (BATCH * HH1 * WW1) / 256;

    // ----- SR pass: features -> srf (f32) -----
    conv1_direct<<<gC1, 256, 0, stream>>>(sr, w1, b1, a1);
    conv3x3_wmma<0><<<gC2, 256, 0, stream>>>(a1, wt2, b2, a2, nullptr, nullptr,
                                             HH1, WW1, CC1, CC2);
    maxpool2x2<<<gP, 256, 0, stream>>>(a2, a2p, CC2);
    conv3x3_wmma<0><<<gC3, 256, 0, stream>>>(a2p, wt3, b3, a3, nullptr, nullptr,
                                             HH2, WW2, CC2, CC3);
    conv3x3_wmma<1><<<gC4, 256, 0, stream>>>(a3, wt4, b4, nullptr, srf, nullptr,
                                             HH2, WW2, CC3, CC4);

    // ----- HR pass: features, fused squared-diff vs srf -> accum[2] -----
    conv1_direct<<<gC1, 256, 0, stream>>>(hr, w1, b1, a1);
    conv3x3_wmma<0><<<gC2, 256, 0, stream>>>(a1, wt2, b2, a2, nullptr, nullptr,
                                             HH1, WW1, CC1, CC2);
    maxpool2x2<<<gP, 256, 0, stream>>>(a2, a2p, CC2);
    conv3x3_wmma<0><<<gC3, 256, 0, stream>>>(a2p, wt3, b3, a3, nullptr, nullptr,
                                             HH2, WW2, CC2, CC3);
    conv3x3_wmma<2><<<gC4, 256, 0, stream>>>(a3, wt4, b4, nullptr, srf, accum + 2,
                                             HH2, WW2, CC3, CC4);

    finalize<<<1, 32, 0, stream>>>(accum, out);
}
